// MultiHeadAtten_8504035246502
// MI455X (gfx1250) — matrix operand order repaired and checked by reading, not compiled, not run on hardware
//
#include <hip/hip_runtime.h>
#include <cstdint>
#include <cstddef>

// ---------------- CDNA5 WMMA types ----------------
typedef __attribute__((ext_vector_type(16))) __bf16 v16bf;
typedef __attribute__((ext_vector_type(8)))  float  v8f;
typedef __attribute__((ext_vector_type(8)))  unsigned short u16x8;

union ABFrag { unsigned int u[8]; v16bf v; };

static constexpr int Bb  = 2;
static constexpr int S   = 2048;
static constexpr int DIN = 1024;
static constexpr int H   = 16;
static constexpr int DH  = 64;
static constexpr int M_TOT = Bb * S;          // 4096 rows of X / output

__device__ __forceinline__ unsigned short f2bf(float f) {
  unsigned int u = __float_as_uint(f);
  u += 0x7FFFu + ((u >> 16) & 1u);            // round-to-nearest-even
  return (unsigned short)(u >> 16);
}

// A fragment (16x32 bf16): lane holds row M=lane%16.
// K map per ISA: VGPR v<4 -> K = (lane/16)*8 + 2*(v%4); v>=4 -> +16. Pairs contiguous.
__device__ __forceinline__ void load_a_frag(ABFrag& a, const unsigned short* tile,
                                            int ldk, int kBase, int lane) {
  const int m  = lane & 15;
  const int hi = (lane >> 4) << 3;            // 0 or 8
  const unsigned short* row = tile + (size_t)m * ldk;
#pragma unroll
  for (int v = 0; v < 8; ++v) {
    int k = kBase + ((v >> 2) << 4) + hi + ((v & 3) << 1);
    a.u[v] = *(const unsigned int*)(row + k);
  }
}

// B fragment (32x16 bf16) from a matrix stored [n][k] (row = output column,
// contiguous K). Lane holds column N=lane%16; K = (lane/16)*8 + 2*(v%4), v>=4 -> +16. Pairs contiguous.
__device__ __forceinline__ void load_bt_frag(ABFrag& b, const unsigned short* tile,
                                             int ldk, int kBase, int lane) {
  const int n  = lane & 15;
  const int hi = (lane >> 4) << 3;            // 0 or 8
  const unsigned short* row = tile + (size_t)n * ldk;
#pragma unroll
  for (int v = 0; v < 8; ++v) {
    int k = kBase + ((v >> 2) << 4) + hi + ((v & 3) << 1);
    b.u[v] = *(const unsigned int*)(row + k);
  }
}

// ---------------- conversion kernels ----------------
__global__ void cvt_x_kernel(const float* __restrict__ x, unsigned short* __restrict__ xb, int n) {
  int i = blockIdx.x * blockDim.x + threadIdx.x;
  if (i < n) xb[i] = f2bf(x[i]);
}

// kernel (3,H,DIN,DH) f32 -> Wt[(p*H+h)*DH + d][k] bf16  (row index == GEMM column n)
__global__ void cvt_wqkv_kernel(const float* __restrict__ wk, unsigned short* __restrict__ wt, int n) {
  int i = blockIdx.x * blockDim.x + threadIdx.x;
  if (i < n) {
    int ph  = i >> 16;          // 64*1024 elements per (p,h)
    int rem = i & 65535;
    int d   = rem >> 10;
    int k   = rem & 1023;
    wt[i] = f2bf(wk[(size_t)ph * 65536 + (size_t)k * 64 + d]);
  }
}

// WO (DIN, DIN) f32 -> WOt[n][f] bf16 (transposed)
__global__ void cvt_wo_kernel(const float* __restrict__ wo, unsigned short* __restrict__ wot, int n) {
  int i = blockIdx.x * blockDim.x + threadIdx.x;
  if (i < n) {
    int nn = i >> 10;
    int f  = i & 1023;
    wot[i] = f2bf(wo[(size_t)f * 1024 + nn]);
  }
}

// ---------------- QKV projection GEMM ----------------
// C(4096 x 3072) = Xb @ Wt^T ; one wave per 64x64 tile (16 accumulators,
// 4 A-frags x 4 B-frags per K step -> 4x frag reuse). Q/K stored [s][d],
// V stored transposed [d][s] so attention's P@V B-frags are contiguous.
__global__ __launch_bounds__(256) void gemm_qkv_kernel(const unsigned short* __restrict__ Xb,
                                                       const unsigned short* __restrict__ Wt,
                                                       unsigned short* __restrict__ qkv) {
  const int lane = threadIdx.x & 31;
  const int w  = blockIdx.x * 8 + (threadIdx.x >> 5);
  const int mt = w % 64;                      // 64 M-tiles of 64 rows
  const int nt = w / 64;                      // 48 N-tiles of 64 cols
  const unsigned short* at = Xb + (size_t)(mt * 64) * DIN;
  const unsigned short* bt = Wt + (size_t)(nt * 64) * DIN;

  v8f acc[16];
#pragma unroll
  for (int i = 0; i < 16; ++i) acc[i] = {};

  for (int k = 0; k < DIN; k += 32) {
    ABFrag a[4], b[4];
#pragma unroll
    for (int i = 0; i < 4; ++i) load_a_frag(a[i], at + (size_t)(i * 16) * DIN, DIN, k, lane);
#pragma unroll
    for (int j = 0; j < 4; ++j) load_bt_frag(b[j], bt + (size_t)(j * 16) * DIN, DIN, k, lane);
#pragma unroll
    for (int i = 0; i < 4; ++i)
#pragma unroll
      for (int j = 0; j < 4; ++j)
        acc[i * 4 + j] = __builtin_amdgcn_wmma_f32_16x16x32_bf16(
            false, a[i].v, false, b[j].v, (short)0, acc[i * 4 + j], false, false);
  }

  const int hi8 = (lane >> 4) << 3;
#pragma unroll
  for (int j = 0; j < 4; ++j) {
    const int ncol = nt * 64 + j * 16 + (lane & 15);
    const int p = ncol >> 10;
    const int h = (ncol >> 6) & 15;
    const int d = ncol & 63;
#pragma unroll
    for (int i = 0; i < 4; ++i) {
      const int rowbase = mt * 64 + i * 16 + hi8;
      const int bidx = rowbase >> 11;         // / S (tiles never straddle batches)
      const int s0 = rowbase & (S - 1);
      v8f c = acc[i * 4 + j];
      if (p < 2) {                            // Q, K: row-major [s][d]
        unsigned short* dst = qkv + ((size_t)((p * Bb + bidx) * H + h)) * S * DH + d;
#pragma unroll
        for (int r = 0; r < 8; ++r) dst[(size_t)(s0 + r) * DH] = f2bf(c[r]);
      } else {                                // V: transposed [d][s], one b128 store
        u16x8 pk;
#pragma unroll
        for (int r = 0; r < 8; ++r) pk[r] = f2bf(c[r]);
        *(u16x8*)(qkv + ((size_t)((2 * Bb + bidx) * H + h)) * S * DH + (size_t)d * S + s0) = pk;
      }
    }
  }
}

// ---------------- flash attention ----------------
// One wave per (b, h, 16-row Q block); online softmax over 64 key-blocks of 32.
__global__ __launch_bounds__(256) void attn_kernel(const unsigned short* __restrict__ qkv,
                                                   unsigned short* __restrict__ ob) {
  __shared__ unsigned short pbuf[8][16][32];  // per-wave P staging (C-layout -> A-layout)
  const int lane = threadIdx.x & 31;
  const int wv   = threadIdx.x >> 5;
  const int w    = blockIdx.x * 8 + wv;
  const int qb   = w & 127;                   // S/16 q-blocks
  const int h    = (w >> 7) & 15;
  const int b    = w >> 11;

  const size_t headStride = (size_t)S * DH;
  const unsigned short* qp  = qkv + (size_t)((0 * Bb + b) * H + h) * headStride; // [s][d]
  const unsigned short* kp  = qkv + (size_t)((1 * Bb + b) * H + h) * headStride; // [s][d]
  const unsigned short* vtp = qkv + (size_t)((2 * Bb + b) * H + h) * headStride; // [d][s]

  ABFrag aq0, aq1;
  load_a_frag(aq0, qp + (size_t)(qb * 16) * DH, DH, 0, lane);
  load_a_frag(aq1, qp + (size_t)(qb * 16) * DH, DH, 32, lane);

  v8f o0 = {}, o1 = {}, o2 = {}, o3 = {};
  float mi[8], li[8];
#pragma unroll
  for (int r = 0; r < 8; ++r) { mi[r] = -3.0e38f; li[r] = 0.f; }

  const float SC = 1.0f / 65536.0f;           // 1/(DH*DH*H)

  for (int kb = 0; kb < S / 32; ++kb) {
    v8f s0 = {}, s1 = {};
    {
      ABFrag bk;
      load_bt_frag(bk, kp + (size_t)(kb * 32) * DH, DH, 0, lane);
      s0 = __builtin_amdgcn_wmma_f32_16x16x32_bf16(false, aq0.v, false, bk.v, (short)0, s0, false, false);
      load_bt_frag(bk, kp + (size_t)(kb * 32) * DH, DH, 32, lane);
      s0 = __builtin_amdgcn_wmma_f32_16x16x32_bf16(false, aq1.v, false, bk.v, (short)0, s0, false, false);
      load_bt_frag(bk, kp + (size_t)(kb * 32 + 16) * DH, DH, 0, lane);
      s1 = __builtin_amdgcn_wmma_f32_16x16x32_bf16(false, aq0.v, false, bk.v, (short)0, s1, false, false);
      load_bt_frag(bk, kp + (size_t)(kb * 32 + 16) * DH, DH, 32, lane);
      s1 = __builtin_amdgcn_wmma_f32_16x16x32_bf16(false, aq1.v, false, bk.v, (short)0, s1, false, false);
    }
    // online softmax stats; C-frag row r lives in one 16-lane half -> shfl_xor 1,2,4,8
    float t[8], p0[8], p1[8], rs[8];
#pragma unroll
    for (int r = 0; r < 8; ++r) {
      p0[r] = s0[r] * SC; p1[r] = s1[r] * SC;
      t[r] = fmaxf(p0[r], p1[r]);
    }
#pragma unroll
    for (int mk = 1; mk < 16; mk <<= 1)
#pragma unroll
      for (int r = 0; r < 8; ++r) t[r] = fmaxf(t[r], __shfl_xor(t[r], mk, 32));
#pragma unroll
    for (int r = 0; r < 8; ++r) {
      float mn   = fmaxf(mi[r], t[r]);
      float corr = __expf(mi[r] - mn);
      mi[r] = mn; li[r] *= corr;
      o0[r] *= corr; o1[r] *= corr; o2[r] *= corr; o3[r] *= corr;
      p0[r] = __expf(p0[r] - mn);
      p1[r] = __expf(p1[r] - mn);
      rs[r] = p0[r] + p1[r];
    }
#pragma unroll
    for (int mk = 1; mk < 16; mk <<= 1)
#pragma unroll
      for (int r = 0; r < 8; ++r) rs[r] += __shfl_xor(rs[r], mk, 32);
#pragma unroll
    for (int r = 0; r < 8; ++r) li[r] += rs[r];

    // P: C-layout -> LDS (row-major 16x32 bf16) -> A-layout reload
    {
      const int col = lane & 15;
      const int rb  = (lane >> 4) << 3;
#pragma unroll
      for (int r = 0; r < 8; ++r) {
        pbuf[wv][rb + r][col]      = f2bf(p0[r]);
        pbuf[wv][rb + r][col + 16] = f2bf(p1[r]);
      }
    }
    asm volatile("s_wait_dscnt 0" ::: "memory");
    ABFrag ap;
    {
      const int m  = lane & 15;
      const int hi = (lane >> 4) << 3;
#pragma unroll
      for (int v = 0; v < 8; ++v) {
        int k = ((v >> 2) << 4) + hi + ((v & 3) << 1);
        ap.u[v] = *(const unsigned int*)&pbuf[wv][m][k];
      }
    }
    {
      ABFrag bv;                               // V^T rows are d, contiguous keys
      load_bt_frag(bv, vtp + (size_t)( 0) * S + kb * 32, S, 0, lane);
      o0 = __builtin_amdgcn_wmma_f32_16x16x32_bf16(false, ap.v, false, bv.v, (short)0, o0, false, false);
      load_bt_frag(bv, vtp + (size_t)(16) * S + kb * 32, S, 0, lane);
      o1 = __builtin_amdgcn_wmma_f32_16x16x32_bf16(false, ap.v, false, bv.v, (short)0, o1, false, false);
      load_bt_frag(bv, vtp + (size_t)(32) * S + kb * 32, S, 0, lane);
      o2 = __builtin_amdgcn_wmma_f32_16x16x32_bf16(false, ap.v, false, bv.v, (short)0, o2, false, false);
      load_bt_frag(bv, vtp + (size_t)(48) * S + kb * 32, S, 0, lane);
      o3 = __builtin_amdgcn_wmma_f32_16x16x32_bf16(false, ap.v, false, bv.v, (short)0, o3, false, false);
    }
  }

  // normalize + write in d-major / H-minor feature order: f = d*16 + h
  const int n  = lane & 15;
  const int rb = (lane >> 4) << 3;
#pragma unroll
  for (int r = 0; r < 8; ++r) {
    float inv = 1.0f / li[r];
    int s = qb * 16 + rb + r;
    size_t rowoff = (size_t)(b * S + s) * (H * DH);
    ob[rowoff + (size_t)(( 0 + n) * 16 + h)] = f2bf(o0[r] * inv);
    ob[rowoff + (size_t)((16 + n) * 16 + h)] = f2bf(o1[r] * inv);
    ob[rowoff + (size_t)((32 + n) * 16 + h)] = f2bf(o2[r] * inv);
    ob[rowoff + (size_t)((48 + n) * 16 + h)] = f2bf(o3[r] * inv);
  }
}

// ---------------- output projection GEMM ----------------
// (4096 x 1024) = OB @ WOt^T ; one wave per 64x64 tile.
__global__ __launch_bounds__(256) void gemm_out_kernel(const unsigned short* __restrict__ A,
                                                       const unsigned short* __restrict__ Bt,
                                                       float* __restrict__ out) {
  const int lane = threadIdx.x & 31;
  const int w  = blockIdx.x * 8 + (threadIdx.x >> 5);
  const int mt = w & 63;                      // 64 M-tiles of 64 rows
  const int nt = w >> 6;                      // 16 N-tiles of 64 cols
  const unsigned short* at = A  + (size_t)(mt * 64) * DIN;
  const unsigned short* bt = Bt + (size_t)(nt * 64) * DIN;

  v8f acc[16];
#pragma unroll
  for (int i = 0; i < 16; ++i) acc[i] = {};

  for (int k = 0; k < DIN; k += 32) {
    ABFrag a[4], b[4];
#pragma unroll
    for (int i = 0; i < 4; ++i) load_a_frag(a[i], at + (size_t)(i * 16) * DIN, DIN, k, lane);
#pragma unroll
    for (int j = 0; j < 4; ++j) load_bt_frag(b[j], bt + (size_t)(j * 16) * DIN, DIN, k, lane);
#pragma unroll
    for (int i = 0; i < 4; ++i)
#pragma unroll
      for (int j = 0; j < 4; ++j)
        acc[i * 4 + j] = __builtin_amdgcn_wmma_f32_16x16x32_bf16(
            false, a[i].v, false, b[j].v, (short)0, acc[i * 4 + j], false, false);
  }

  const int hi8 = (lane >> 4) << 3;
#pragma unroll
  for (int j = 0; j < 4; ++j) {
    const int ncol = nt * 64 + j * 16 + (lane & 15);
#pragma unroll
    for (int i = 0; i < 4; ++i) {
      const int rowbase = mt * 64 + i * 16 + hi8;
      v8f c = acc[i * 4 + j];
#pragma unroll
      for (int r = 0; r < 8; ++r)
        out[(size_t)(rowbase + r) * DIN + ncol] = c[r];
    }
  }
}

// ---------------- launch ----------------
extern "C" void kernel_launch(void* const* d_in, const int* in_sizes, int n_in,
                              void* d_out, int out_size, void* d_ws, size_t ws_size,
                              hipStream_t stream) {
  (void)in_sizes; (void)n_in; (void)out_size; (void)ws_size;
  const float* x  = (const float*)d_in[0];
  const float* wk = (const float*)d_in[1];
  const float* wo = (const float*)d_in[2];
  float* out = (float*)d_out;

  // workspace layout (bf16 = unsigned short), ~48 MiB total
  unsigned short* Xb  = (unsigned short*)d_ws;
  unsigned short* Wt  = Xb  + (size_t)M_TOT * DIN;        // 4,194,304
  unsigned short* WOt = Wt  + (size_t)3 * H * DH * DIN;   // 3,145,728
  unsigned short* QKV = WOt + (size_t)DIN * DIN;          // 1,048,576
  unsigned short* OB  = QKV + (size_t)3 * Bb * H * S * DH;// 12,582,912

  cvt_x_kernel   <<<(M_TOT * DIN) / 256, 256, 0, stream>>>(x,  Xb,  M_TOT * DIN);
  cvt_wqkv_kernel<<<(3 * H * DIN * DH) / 256, 256, 0, stream>>>(wk, Wt, 3 * H * DIN * DH);
  cvt_wo_kernel  <<<(DIN * DIN) / 256, 256, 0, stream>>>(wo, WOt, DIN * DIN);

  gemm_qkv_kernel<<<(64 * 48) / 8, 256, 0, stream>>>(Xb, Wt, QKV);
  attn_kernel    <<<(Bb * H * (S / 16)) / 8, 256, 0, stream>>>(QKV, OB);
  gemm_out_kernel<<<(64 * 16) / 8, 256, 0, stream>>>(OB, WOt, out);
}
